// ScalingAttention_74105365725832
// MI455X (gfx1250) — matrix operand order repaired and checked by reading, not compiled, not run on hardware
//
#include <hip/hip_runtime.h>

// ---------------------------------------------------------------------------
// ScalingAttention for MI455X (gfx1250, wave32, WMMA 16x16x32 f16->f32)
// Software-pipelined: double-buffered LDS, async global->LDS (ASYNCcnt),
// register-staged f32 weight tiles, one barrier per tile iteration.
// ---------------------------------------------------------------------------

#define B_   2
#define NQ_  4096
#define NKV_ 1024
#define D_   512
#define H_   8
#define DH_  64

#define TRY_ASYNC 1

typedef _Float16 f16;
typedef __attribute__((ext_vector_type(2)))  _Float16 f16x2;
typedef __attribute__((ext_vector_type(4)))  _Float16 f16x4;
typedef __attribute__((ext_vector_type(8)))  _Float16 f16x8;
typedef __attribute__((ext_vector_type(16))) _Float16 v16h;
typedef __attribute__((ext_vector_type(8)))  float    v8f;
typedef __attribute__((ext_vector_type(4)))  int      i32x4;

__device__ __forceinline__ v8f wmma_f16(v16h a, v16h b, v8f c) {
  return __builtin_amdgcn_wmma_f32_16x16x32_f16(
      false, a, false, b, (short)0, c, false, false);
}

__device__ __forceinline__ v8f v8f_zero() {
  v8f z = {0.f, 0.f, 0.f, 0.f, 0.f, 0.f, 0.f, 0.f};
  return z;
}

// --------- CDNA5 async global->LDS copy (ASYNCcnt-tracked) -----------------
#if TRY_ASYNC && defined(__gfx1250__) && \
    __has_builtin(__builtin_amdgcn_global_load_async_to_lds_b128)
#define USE_ASYNC_LDS 1
__device__ __forceinline__ void async_copy_b128(const f16* g, f16* l) {
  __builtin_amdgcn_global_load_async_to_lds_b128(
      (__attribute__((address_space(1))) i32x4*)g,
      (__attribute__((address_space(3))) i32x4*)l, 0, 0);
}
__device__ __forceinline__ void async_wait0() {
#if __has_builtin(__builtin_amdgcn_s_wait_asynccnt)
  __builtin_amdgcn_s_wait_asynccnt(0);
#else
  asm volatile("s_wait_asynccnt 0" ::: "memory");
#endif
}
#else
#define USE_ASYNC_LDS 0
__device__ __forceinline__ void async_copy_b128(const f16* g, f16* l) {
  *(f16x8*)l = *(const f16x8*)g;
}
__device__ __forceinline__ void async_wait0() {}
#endif

// 16-lane row reductions (rows of a 16x16 C tile live in lane halves).
__device__ __forceinline__ float row_red_max(float x) {
  x = fmaxf(x, __shfl_xor(x, 1, 32));
  x = fmaxf(x, __shfl_xor(x, 2, 32));
  x = fmaxf(x, __shfl_xor(x, 4, 32));
  x = fmaxf(x, __shfl_xor(x, 8, 32));
  return x;
}
__device__ __forceinline__ float row_red_sum(float x) {
  x += __shfl_xor(x, 1, 32);
  x += __shfl_xor(x, 2, 32);
  x += __shfl_xor(x, 4, 32);
  x += __shfl_xor(x, 8, 32);
  return x;
}

// A-fragment (16x32 f16, MxK): lane l holds row m=l&15, khalf=l>>4.
// element e -> k = (e&7) + 8*kh + 16*(e>>3)  => pair p -> (p&3)+4*kh+8*(p>>2)
__device__ __forceinline__ v16h load_afrag_lds(const f16 (*lds)[32], int ra, int lane) {
  const int m   = ra + (lane & 15);
  const int kh4 = (lane >> 4) * 4;
  const f16x2* row = (const f16x2*)lds[m];
  v16h a;
#pragma unroll
  for (int p = 0; p < 8; ++p) {
    f16x2 d = row[(p & 3) + kh4 + ((p >> 2) << 3)];
    a[2 * p] = d.x; a[2 * p + 1] = d.y;
  }
  return a;
}

// B-fragment (32x16 f16, KxN) from a [N][32] (contraction-contiguous) tile:
// lane l holds col n=l&15; element e -> k = (e&7) + 8*kh + 16*(e>>3) => pair p -> (p&3)+4*kh+8*(p>>2)
__device__ __forceinline__ v16h load_bfrag_lds(const f16 (*ldsT)[32], int nb, int lane) {
  const int n   = nb + (lane & 15);
  const int kh4 = (lane >> 4) * 4;
  const f16x2* row = (const f16x2*)ldsT[n];
  v16h b;
#pragma unroll
  for (int p = 0; p < 8; ++p) {
    f16x2 d = row[(p & 3) + kh4 + ((p >> 2) << 3)];
    b[2 * p] = d.x; b[2 * p + 1] = d.y;
  }
  return b;
}

// ---------------------------------------------------------------------------
// proj_kernel: Cf16[M,512] = f16(A_f32[M,512] @ W_f32[512,512])
// Pipelined: next A/W tiles live in registers across the WMMA block.
// ---------------------------------------------------------------------------
__global__ __launch_bounds__(256)
void proj_kernel(const float* __restrict__ A, const float* __restrict__ W,
                 f16* __restrict__ C) {
  __shared__ f16 ldsA[2][128][32];   // [m][k]
  __shared__ f16 ldsWt[2][128][32];  // [n][k]

  const int tid = threadIdx.x, lane = tid & 31, wv = tid >> 5;
  const int wr = wv >> 1, wc = wv & 1;
  const int m0 = blockIdx.y * 128, n0 = blockIdx.x * 128;

  float4 aReg[4], wReg[4];
  auto load_regs = [&](int k0) {
#pragma unroll
    for (int t = 0; t < 4; ++t) {
      const int idx = tid + t * 256;
      const int r = idx >> 3, c4 = idx & 7;          // A: 128 rows x 8 float4
      aReg[t] = *(const float4*)(A + (size_t)(m0 + r) * 512 + k0 + c4 * 4);
      const int kr = idx >> 5, w4 = idx & 31;        // W: 32 rows x 32 float4
      wReg[t] = *(const float4*)(W + (size_t)(k0 + kr) * 512 + n0 + w4 * 4);
    }
  };
  auto store_lds = [&](int buf) {
#pragma unroll
    for (int t = 0; t < 4; ++t) {
      const int idx = tid + t * 256;
      const int r = idx >> 3, c4 = idx & 7;
      f16x4 h; h.x = (f16)aReg[t].x; h.y = (f16)aReg[t].y;
      h.z = (f16)aReg[t].z; h.w = (f16)aReg[t].w;
      *(f16x4*)&ldsA[buf][r][c4 * 4] = h;
      const int kr = idx >> 5, w4 = idx & 31;
      ldsWt[buf][w4 * 4 + 0][kr] = (f16)wReg[t].x;
      ldsWt[buf][w4 * 4 + 1][kr] = (f16)wReg[t].y;
      ldsWt[buf][w4 * 4 + 2][kr] = (f16)wReg[t].z;
      ldsWt[buf][w4 * 4 + 3][kr] = (f16)wReg[t].w;
    }
  };

  v8f acc[2][4];
#pragma unroll
  for (int i = 0; i < 2; ++i)
#pragma unroll
    for (int j = 0; j < 4; ++j) acc[i][j] = v8f_zero();

  load_regs(0);
  store_lds(0);
  __syncthreads();

  for (int it = 0; it < 16; ++it) {
    const int buf = it & 1;
    if (it + 1 < 16) load_regs((it + 1) * 32);   // global loads overlap WMMAs

    v16h af[2], bf[4];
#pragma unroll
    for (int rf = 0; rf < 2; ++rf)
      af[rf] = load_afrag_lds(ldsA[buf], wr * 32 + rf * 16, lane);
#pragma unroll
    for (int cf = 0; cf < 4; ++cf)
      bf[cf] = load_bfrag_lds(ldsWt[buf], wc * 64 + cf * 16, lane);
#pragma unroll
    for (int rf = 0; rf < 2; ++rf)
#pragma unroll
      for (int cf = 0; cf < 4; ++cf)
        acc[rf][cf] = wmma_f16(af[rf], bf[cf], acc[rf][cf]);

    if (it + 1 < 16) store_lds(buf ^ 1);
    __syncthreads();
  }

  const int kh = lane >> 4, nl = lane & 15;
#pragma unroll
  for (int rf = 0; rf < 2; ++rf)
#pragma unroll
    for (int cf = 0; cf < 4; ++cf) {
      const int gmb = m0 + wr * 32 + rf * 16 + 8 * kh;
      const int gn  = n0 + wc * 64 + cf * 16 + nl;
#pragma unroll
      for (int r = 0; r < 8; ++r)
        C[(size_t)(gmb + r) * 512 + gn] = (f16)acc[rf][cf][r];
    }
}

// ---------------------------------------------------------------------------
// out_kernel: C32[M,512] = A_f16[M,512] @ W_f32[512,512] + bias
// Pipelined: A via async global->LDS into alternate buffer; W via registers.
// ---------------------------------------------------------------------------
__global__ __launch_bounds__(256)
void out_kernel(const f16* __restrict__ A, const float* __restrict__ W,
                const float* __restrict__ bias, float* __restrict__ C) {
  __shared__ f16 ldsA[2][128][32];
  __shared__ f16 ldsWt[2][128][32];

  const int tid = threadIdx.x, lane = tid & 31, wv = tid >> 5;
  const int wr = wv >> 1, wc = wv & 1;
  const int m0 = blockIdx.y * 128, n0 = blockIdx.x * 128;

  auto issue_a = [&](int k0, int buf) {
#pragma unroll
    for (int t = 0; t < 2; ++t) {
      const int idx = tid + t * 256;       // 512 b128 total
      const int r = idx >> 2, c = (idx & 3) * 8;
      async_copy_b128(A + (size_t)(m0 + r) * 512 + k0 + c, &ldsA[buf][r][c]);
    }
  };
  float4 wReg[4];
  auto load_w = [&](int k0) {
#pragma unroll
    for (int t = 0; t < 4; ++t) {
      const int idx = tid + t * 256;
      const int kr = idx >> 5, w4 = idx & 31;
      wReg[t] = *(const float4*)(W + (size_t)(k0 + kr) * 512 + n0 + w4 * 4);
    }
  };
  auto store_w = [&](int buf) {
#pragma unroll
    for (int t = 0; t < 4; ++t) {
      const int idx = tid + t * 256;
      const int kr = idx >> 5, w4 = idx & 31;
      ldsWt[buf][w4 * 4 + 0][kr] = (f16)wReg[t].x;
      ldsWt[buf][w4 * 4 + 1][kr] = (f16)wReg[t].y;
      ldsWt[buf][w4 * 4 + 2][kr] = (f16)wReg[t].z;
      ldsWt[buf][w4 * 4 + 3][kr] = (f16)wReg[t].w;
    }
  };

  v8f acc[2][4];
#pragma unroll
  for (int i = 0; i < 2; ++i)
#pragma unroll
    for (int j = 0; j < 4; ++j) acc[i][j] = v8f_zero();

  issue_a(0, 0);
  load_w(0);
  store_w(0);
  async_wait0();
  __syncthreads();

  for (int it = 0; it < 16; ++it) {
    const int buf = it & 1;
    const bool more = (it + 1 < 16);
    if (more) { issue_a((it + 1) * 32, buf ^ 1); load_w((it + 1) * 32); }

    v16h af[2], bf[4];
#pragma unroll
    for (int rf = 0; rf < 2; ++rf)
      af[rf] = load_afrag_lds(ldsA[buf], wr * 32 + rf * 16, lane);
#pragma unroll
    for (int cf = 0; cf < 4; ++cf)
      bf[cf] = load_bfrag_lds(ldsWt[buf], wc * 64 + cf * 16, lane);
#pragma unroll
    for (int rf = 0; rf < 2; ++rf)
#pragma unroll
      for (int cf = 0; cf < 4; ++cf)
        acc[rf][cf] = wmma_f16(af[rf], bf[cf], acc[rf][cf]);

    if (more) { store_w(buf ^ 1); async_wait0(); }
    __syncthreads();
  }

  const int kh = lane >> 4, nl = lane & 15;
#pragma unroll
  for (int rf = 0; rf < 2; ++rf)
#pragma unroll
    for (int cf = 0; cf < 4; ++cf) {
      const int gmb = m0 + wr * 32 + rf * 16 + 8 * kh;
      const int gn  = n0 + wc * 64 + cf * 16 + nl;
      const float bv = bias[gn];
#pragma unroll
      for (int r = 0; r < 8; ++r)
        C[(size_t)(gmb + r) * 512 + gn] = acc[rf][cf][r] + bv;
    }
}

// ---------------------------------------------------------------------------
// Block-causal flash attention, pipelined KV chunks of 32 keys.
// Block = 8 waves; wave w owns 16 queries. Key j allowed iff j <= i>>2.
// K chunk: async global->LDS; V chunk: register pipeline (transposed store).
// ---------------------------------------------------------------------------
__global__ __launch_bounds__(256)
void attn_kernel(const f16* __restrict__ Q, const f16* __restrict__ K,
                 const f16* __restrict__ V, float* __restrict__ O) {
  __shared__ f16 ldsK[2][32][64];    // [kv][d]
  __shared__ f16 ldsVt[2][64][32];   // [d][kv]
  __shared__ f16 ldsP[8][16][32];    // per-wave P staging (C- -> A-layout)

  const int tid = threadIdx.x, lane = tid & 31, wv = tid >> 5;
  const int h = blockIdx.y, b = blockIdx.z;
  const int qblk = blockIdx.x * 128;
  const int qi0  = qblk + wv * 16;
  const int kh = lane >> 4, nl = lane & 15;
  const int krow = tid >> 3, kc8 = (tid & 7) * 8;

  auto issue_k = [&](int kb, int buf) {
    async_copy_b128(K + (size_t)(b * NKV_ + kb + krow) * D_ + h * DH_ + kc8,
                    &ldsK[buf][krow][kc8]);
  };
  f16x2 vReg[4];
  auto load_v = [&](int kb) {
#pragma unroll
    for (int t = 0; t < 4; ++t) {
      const int idx = tid + t * 256;
      const int kv = idx >> 5, dp = idx & 31;
      vReg[t] = *(const f16x2*)(V + (size_t)(b * NKV_ + kb + kv) * D_ +
                                h * DH_ + 2 * dp);
    }
  };
  auto store_v = [&](int buf) {
#pragma unroll
    for (int t = 0; t < 4; ++t) {
      const int idx = tid + t * 256;
      const int kv = idx >> 5, dp = idx & 31;
      ldsVt[buf][2 * dp][kv]     = vReg[t].x;
      ldsVt[buf][2 * dp + 1][kv] = vReg[t].y;
    }
  };

  // Q A-fragments (16 rows x DH=64 = 2 k-chunks) gathered from global.
  v16h qf[2];
  {
    const f16x2* qp =
        (const f16x2*)(Q + ((size_t)(b * NQ_ + qi0 + nl) * D_ + h * DH_));
    const int kh4 = kh * 4;
#pragma unroll
    for (int c = 0; c < 2; ++c)
#pragma unroll
      for (int p = 0; p < 8; ++p) {
        f16x2 d = qp[c * 16 + (p & 3) + kh4 + ((p >> 2) << 3)];
        qf[c][2 * p] = d.x; qf[c][2 * p + 1] = d.y;
      }
  }

  v8f oacc[4];
#pragma unroll
  for (int j = 0; j < 4; ++j) oacc[j] = v8f_zero();
  float mrun[8], lrun[8];
#pragma unroll
  for (int r = 0; r < 8; ++r) { mrun[r] = -1e30f; lrun[r] = 0.f; }

  const int kmax_w  = (qi0 + 15) >> 2;                 // last allowed key
  const int nchunks = (((qblk + 127) >> 2) >> 5) + 1;  // block-wide bound

  issue_k(0, 0);
  load_v(0);
  store_v(0);
  async_wait0();
  __syncthreads();

  for (int c = 0; c < nchunks; ++c) {
    const int buf = c & 1;
    const int kb = c * 32;
    const bool more = (c + 1 < nchunks);
    if (more) { issue_k(kb + 32, buf ^ 1); load_v(kb + 32); }
    if (c + 2 < nchunks) {   // prefetch chunk c+2 (global_prefetch_b8)
      const size_t nb2 = (size_t)(b * NKV_ + kb + 64 + krow) * D_ + h * DH_;
      __builtin_prefetch(K + nb2 + kc8, 0, 1);
      __builtin_prefetch(V + nb2 + kc8, 0, 1);
    }

    // Wave-uniform activity predicate (scalar branch keeps EXEC all-1s).
    const int act = __builtin_amdgcn_readfirstlane((kb <= kmax_w) ? 1 : 0);
    if (act) {
      // S = (Q K^T) * DH^-0.5: two 16-key halves, contraction over DH=64.
      v8f s[2];
#pragma unroll
      for (int t = 0; t < 2; ++t) {
        s[t] = v8f_zero();
        const f16x2* krw = (const f16x2*)ldsK[buf][16 * t + nl];
        v16h kfa, kfb;
#pragma unroll
        for (int p = 0; p < 8; ++p) {
          f16x2 d0 = krw[(p & 3) + 4 * kh + ((p >> 2) << 3)];        // d-chunk 0
          f16x2 d1 = krw[16 + (p & 3) + 4 * kh + ((p >> 2) << 3)];   // d-chunk 1
          kfa[2 * p] = d0.x; kfa[2 * p + 1] = d0.y;
          kfb[2 * p] = d1.x; kfb[2 * p + 1] = d1.y;
        }
        s[t] = wmma_f16(qf[0], kfa, s[t]);
        s[t] = wmma_f16(qf[1], kfb, s[t]);
      }
      // Scale + block-causal mask.
#pragma unroll
      for (int t = 0; t < 2; ++t)
#pragma unroll
        for (int r = 0; r < 8; ++r) {
          float val = s[t][r] * 0.125f;
          const int qi = qi0 + r + 8 * kh;
          const int kj = kb + t * 16 + nl;
          if (kj > (qi >> 2)) val = -1e30f;
          s[t][r] = val;
        }
      // Online softmax update.
#pragma unroll
      for (int r = 0; r < 8; ++r) {
        const float cm = row_red_max(fmaxf(s[0][r], s[1][r]));
        const float nm = fmaxf(mrun[r], cm);
        const float alpha = __expf(mrun[r] - nm);
        const float e0 = __expf(s[0][r] - nm);
        const float e1 = __expf(s[1][r] - nm);
        const float rs = row_red_sum(e0 + e1);
        lrun[r] = lrun[r] * alpha + rs;
        mrun[r] = nm;
#pragma unroll
        for (int j = 0; j < 4; ++j) oacc[j][r] *= alpha;
        ldsP[wv][r + 8 * kh][nl]      = (f16)e0;
        ldsP[wv][r + 8 * kh][16 + nl] = (f16)e1;
      }
      // P (C-layout in LDS) -> A-fragment; same-wave LDS ops stay in order.
      v16h pf;
      {
        const f16x2* prow = (const f16x2*)ldsP[wv][nl];
        const int kh4 = kh * 4;
#pragma unroll
        for (int p = 0; p < 8; ++p) {
          f16x2 d = prow[(p & 3) + kh4 + ((p >> 2) << 3)];
          pf[2 * p] = d.x; pf[2 * p + 1] = d.y;
        }
      }
      // O += P V over 4 DH-16 column tiles.
#pragma unroll
      for (int j = 0; j < 4; ++j) {
        v16h vf = load_bfrag_lds(ldsVt[buf], 16 * j, lane);
        oacc[j] = wmma_f16(pf, vf, oacc[j]);
      }
    }

    if (more) { store_v(buf ^ 1); async_wait0(); }
    __syncthreads();
  }

  // Normalize by softmax denominator and write f32 head-slice output.
#pragma unroll
  for (int r = 0; r < 8; ++r) {
    const float inv = 1.0f / lrun[r];
    const int qi = qi0 + r + 8 * kh;
#pragma unroll
    for (int j = 0; j < 4; ++j)
      O[(size_t)(b * NQ_ + qi) * D_ + h * DH_ + 16 * j + nl] = oacc[j][r] * inv;
  }
}

// ---------------------------------------------------------------------------
// LayerNorm over D=512, f32 in -> f16 out. One block (256 threads) per row.
// ---------------------------------------------------------------------------
__global__ __launch_bounds__(256)
void ln_kernel(const float* __restrict__ X, const float* __restrict__ gamma,
               const float* __restrict__ beta, f16* __restrict__ Y) {
  __shared__ float red[16];
  const int row = blockIdx.x, tid = threadIdx.x;
  const float* x = X + (size_t)row * 512;
  const float v0 = x[tid], v1 = x[tid + 256];
  float s = v0 + v1, ss = v0 * v0 + v1 * v1;
#pragma unroll
  for (int off = 1; off < 32; off <<= 1) {
    s  += __shfl_xor(s, off, 32);
    ss += __shfl_xor(ss, off, 32);
  }
  const int w = tid >> 5, lane = tid & 31;
  if (lane == 0) { red[w] = s; red[8 + w] = ss; }
  __syncthreads();
  if (tid == 0) {
    float S = 0.f, SS = 0.f;
    for (int i = 0; i < 8; ++i) { S += red[i]; SS += red[8 + i]; }
    red[0] = S; red[1] = SS;
  }
  __syncthreads();
  const float mean = red[0] * (1.f / 512.f);
  const float var  = red[1] * (1.f / 512.f) - mean * mean;
  const float inv  = rsqrtf(var + 1e-5f);
  Y[(size_t)row * 512 + tid]       = (f16)(((v0 - mean) * inv) * gamma[tid] + beta[tid]);
  Y[(size_t)row * 512 + tid + 256] = (f16)(((v1 - mean) * inv) * gamma[tid + 256] + beta[tid + 256]);
}

// ---------------------------------------------------------------------------
extern "C" void kernel_launch(void* const* d_in, const int* in_sizes, int n_in,
                              void* d_out, int out_size, void* d_ws, size_t ws_size,
                              hipStream_t stream) {
  (void)in_sizes; (void)n_in; (void)out_size; (void)ws_size;
  const float* q     = (const float*)d_in[0];
  const float* kv    = (const float*)d_in[1];
  const float* Wq    = (const float*)d_in[2];
  const float* Wk    = (const float*)d_in[3];
  const float* Wv    = (const float*)d_in[4];
  const float* Wo    = (const float*)d_in[5];
  const float* bo    = (const float*)d_in[6];
  const float* gamma = (const float*)d_in[7];
  const float* beta  = (const float*)d_in[8];
  float* out = (float*)d_out;

  f16*   Qf = (f16*)d_ws;                             // 8 MB (reused as X)
  f16*   Kf = Qf + (size_t)B_ * NQ_ * D_;             // 2 MB
  f16*   Vf = Kf + (size_t)B_ * NKV_ * D_;            // 2 MB
  float* AO = (float*)(Vf + (size_t)B_ * NKV_ * D_);  // 16 MB

  const int MQ  = B_ * NQ_;   // 8192
  const int MKV = B_ * NKV_;  // 2048

  proj_kernel<<<dim3(4, MQ / 128),  256, 0, stream>>>(q,  Wq, Qf);
  proj_kernel<<<dim3(4, MKV / 128), 256, 0, stream>>>(kv, Wk, Kf);
  proj_kernel<<<dim3(4, MKV / 128), 256, 0, stream>>>(kv, Wv, Vf);
  attn_kernel<<<dim3(NQ_ / 128, H_, B_), 256, 0, stream>>>(Qf, Kf, Vf, AO);
  ln_kernel<<<dim3(MQ), 256, 0, stream>>>(AO, gamma, beta, Qf);
  out_kernel<<<dim3(4, MQ / 128),  256, 0, stream>>>(Qf, Wo, bo, out);
}
